// MultiHeadSelfAttention_6803228196918
// MI455X (gfx1250) — matrix operand order repaired and checked by reading, not compiled, not run on hardware
//
#include <hip/hip_runtime.h>
#include <cstddef>

// ---------------------------------------------------------------------------
// MI455X (gfx1250) multi-head self-attention, bf16 WMMA path, f32 accumulate.
// ---------------------------------------------------------------------------

typedef __bf16 bf16;
typedef __attribute__((ext_vector_type(8)))  bf16  v8bf;
typedef __attribute__((ext_vector_type(16))) bf16  v16bf;
typedef __attribute__((ext_vector_type(8)))  float v8f;

#define D_MODEL   2048
#define N_HEADS   16
#define HEAD_DIM  128
#define SEQ       2048
#define BATCH     4
#define BT        (BATCH * SEQ)        // 8192 rows
#define QKV_N     (3 * D_MODEL)        // 6144

// ---- WMMA wrapper: v_wmma_f32_16x16x32_bf16 -------------------------------
__device__ __forceinline__ v8f wmma_bf16(v16bf a, v16bf b, v8f c) {
  // (neg_a, A, neg_b, B, c_mod, C, reuse_a, reuse_b)
  return __builtin_amdgcn_wmma_f32_16x16x32_bf16(false, a, false, b,
                                                 (short)0, c, false, false);
}

// ---- A-matrix fragment: 16(M) x 32(K) bf16, row-major source --------------
// ISA layout: lane m = lane&15 ; kbase = (lane<16)?0:8 ;
//   j<8 -> k = kbase + j ; j>=8 -> k = kbase + 8 + j   (two 8-elem chunks)
__device__ __forceinline__ v16bf load_a_frag(const bf16* __restrict__ base, int ld) {
  const int lane = threadIdx.x & 31;
  const int m    = lane & 15;
  const int kb   = (lane >> 4) * 8;
  const bf16* p  = base + m * ld + kb;
  v8bf lo = *(const v8bf*)(p);
  v8bf hi = *(const v8bf*)(p + 16);
  v16bf r;
#pragma unroll
  for (int j = 0; j < 8; ++j) { r[j] = lo[j]; r[j + 8] = hi[j]; }
  return r;
}

// ---- B-matrix fragment: 32(K) x 16(N) bf16 --------------------------------
// Source is "n-major, k contiguous" (row n holds B[k][n] = src[n][k]).
// ISA layout: lane n = lane&15 ; kbase = (lane<16)?0:8 ;
//   j<8 -> k = kbase + j ; j>=8 -> k = kbase + 8 + j   (two 8-elem chunks)
__device__ __forceinline__ v16bf load_b_frag(const bf16* __restrict__ base, int ld) {
  const int lane = threadIdx.x & 31;
  const int n    = lane & 15;
  const int kb   = (lane >> 4) * 8;
  const bf16* p  = base + n * ld + kb;
  v8bf lo = *(const v8bf*)(p);
  v8bf hi = *(const v8bf*)(p + 16);
  v16bf r;
#pragma unroll
  for (int j = 0; j < 8; ++j) { r[j] = lo[j]; r[j + 8] = hi[j]; }
  return r;
}

// ---- fragment set for a 32x64 per-wave GEMM tile (one 32-wide k-step) -----
struct Frags {
  v16bf a0, a1;   // two 16-row A fragments
  v16bf b[4];     // four 16-col B fragments
};

__device__ __forceinline__ void load_frags(Frags& f, const bf16* __restrict__ A0,
                                           const bf16* __restrict__ A1,
                                           const bf16* __restrict__ Wn0, int k) {
  f.a0 = load_a_frag(A0 + k, D_MODEL);
  f.a1 = load_a_frag(A1 + k, D_MODEL);
#pragma unroll
  for (int nt = 0; nt < 4; ++nt)
    f.b[nt] = load_b_frag(Wn0 + (size_t)nt * 16 * D_MODEL + k, D_MODEL);
}

__device__ __forceinline__ void do_wmmas(v8f acc[2][4], const Frags& f) {
#pragma unroll
  for (int nt = 0; nt < 4; ++nt) {
    acc[0][nt] = wmma_bf16(f.a0, f.b[nt], acc[0][nt]);
    acc[1][nt] = wmma_bf16(f.a1, f.b[nt], acc[1][nt]);
  }
}

// ===========================================================================
// Kernel 1: fp32 -> bf16 cast, 8 elements per thread (b128 in, b128 out)
// ===========================================================================
__global__ void k_cast8(const float4* __restrict__ in, v8bf* __restrict__ out, int n8) {
  int i = blockIdx.x * blockDim.x + threadIdx.x;
  if (i < n8) {
    float4 f0 = in[2 * i], f1 = in[2 * i + 1];
    v8bf o;
    o[0] = (bf16)f0.x; o[1] = (bf16)f0.y; o[2] = (bf16)f0.z; o[3] = (bf16)f0.w;
    o[4] = (bf16)f1.x; o[5] = (bf16)f1.y; o[6] = (bf16)f1.z; o[7] = (bf16)f1.w;
    out[i] = o;
  }
}

// ===========================================================================
// Kernel 2: QKV GEMM  qkv[m, e] = sum_d x[m, d] * w_qkv[e, d]
//   block = 256 threads (8 waves) arranged 2(m) x 4(n); per-wave tile 32x64.
//   Ping-pong k-loop (unroll 2): loads land directly in consumed registers,
//   no inter-buffer copies; WMMAs overlap the other buffer's loads.
//   Epilogue scatters into Q (b,h,t,d), K (b,h,t,d), V^T (b,h,d,t), bf16.
// ===========================================================================
__global__ void __launch_bounds__(256)
k_gemm_qkv(const bf16* __restrict__ X, const bf16* __restrict__ W,
           bf16* __restrict__ Q, bf16* __restrict__ K, bf16* __restrict__ VT) {
  const int wave = threadIdx.x >> 5;
  const int lane = threadIdx.x & 31;
  const int m0   = blockIdx.x * 64 + (wave >> 2) * 32;
  const int n0   = blockIdx.y * 256 + (wave & 3) * 64;

  const bf16* A0  = X + (size_t)m0 * D_MODEL;
  const bf16* A1  = X + (size_t)(m0 + 16) * D_MODEL;
  const bf16* Wn0 = W + (size_t)n0 * D_MODEL;

  v8f acc[2][4] = {};
  Frags ping, pong;
  load_frags(ping, A0, A1, Wn0, 0);

  int k = 0;
  for (; k + 64 < D_MODEL; k += 64) {
    load_frags(pong, A0, A1, Wn0, k + 32);
    do_wmmas(acc, ping);
    load_frags(ping, A0, A1, Wn0, k + 64);
    do_wmmas(acc, pong);
  }
  // tail: k == D_MODEL - 64
  load_frags(pong, A0, A1, Wn0, k + 32);
  do_wmmas(acc, ping);
  do_wmmas(acc, pong);

  const int col    = lane & 15;
  const int rowoff = (lane >> 4) * 8;
#pragma unroll
  for (int mt = 0; mt < 2; ++mt) {
#pragma unroll
    for (int nt = 0; nt < 4; ++nt) {
      const int e     = n0 + nt * 16 + col;
      const int which = e >> 11;          // 0=q, 1=k, 2=v
      const int h     = (e >> 7) & 15;
      const int d     = e & 127;
#pragma unroll
      for (int r = 0; r < 8; ++r) {
        const int m  = m0 + mt * 16 + rowoff + r;
        const int bi = m >> 11;
        const int ti = m & (SEQ - 1);
        const int bh = bi * N_HEADS + h;
        const bf16 v = (bf16)acc[mt][nt][r];
        if (which == 0)      Q [((size_t)bh * SEQ + ti) * HEAD_DIM + d] = v;
        else if (which == 1) K [((size_t)bh * SEQ + ti) * HEAD_DIM + d] = v;
        else                 VT[((size_t)bh * HEAD_DIM + d) * SEQ + ti] = v;
      }
    }
  }
}

// ===========================================================================
// Kernel 3: RoPE in place on Q and K (b,h,t,d) bf16; folds 1/sqrt(128) into Q
//   block = 256 threads = 4 positions x 64 pairs
// ===========================================================================
__global__ void __launch_bounds__(256) k_rope(bf16* __restrict__ Q, bf16* __restrict__ K) {
  const int pos = blockIdx.x * 4 + (threadIdx.x >> 6);  // bh * SEQ + t
  const int t   = pos & (SEQ - 1);
  const int i   = threadIdx.x & 63;                     // pair index 0..63
  // inv_freq = 10000^(-i/64) = exp(-i/64 * ln(10000))
  const float inv_freq = __expf(-(float)i * (9.210340371976184f / 64.0f));
  const float ang = (float)t * inv_freq;
  float s, c;
  __sincosf(ang, &s, &c);
  const float qscale = 0.08838834764831845f;  // 1/sqrt(HEAD_DIM)

  const size_t base = (size_t)pos * HEAD_DIM;
  float q1 = (float)Q[base + i], q2 = (float)Q[base + i + 64];
  Q[base + i]      = (bf16)((q1 * c - q2 * s) * qscale);
  Q[base + i + 64] = (bf16)((q2 * c + q1 * s) * qscale);
  float k1 = (float)K[base + i], k2 = (float)K[base + i + 64];
  K[base + i]      = (bf16)(k1 * c - k2 * s);
  K[base + i + 64] = (bf16)(k2 * c + k1 * s);
}

// ===========================================================================
// Kernel 4: flash attention.
//   grid = (SEQ/64, BATCH*N_HEADS), block = 128 (4 waves).
//   Each wave: 16 query rows, Q fragments register-resident, 32-key tiles.
//   K fragments are loop-carried: the NEXT tile's K loads and this tile's V
//   loads are both issued right after the S WMMAs, so their latency hides
//   under the softmax VALU chain and the PV WMMAs. P staging uses per-wave
//   LDS + wave_barrier only (same-wave LDS ops execute in order).
// ===========================================================================
__global__ void __launch_bounds__(128)
k_attn(const bf16* __restrict__ Q, const bf16* __restrict__ Kmat,
       const bf16* __restrict__ VT, const unsigned char* __restrict__ mask,
       bf16* __restrict__ OUT) {
  __shared__ alignas(16) bf16 p_lds[4][16 * 32];   // per-wave P staging (4 KB)

  const int wave = threadIdx.x >> 5;
  const int lane = threadIdx.x & 31;
  const int bh   = blockIdx.y;
  const int bi   = bh >> 4;
  const int h    = bh & 15;
  const int q0   = blockIdx.x * 64 + wave * 16;

  const bf16* Qb = Q    + ((size_t)bh * SEQ + q0) * HEAD_DIM;
  const bf16* Kb = Kmat + (size_t)bh * SEQ * HEAD_DIM;
  const bf16* Vb = VT   + (size_t)bh * HEAD_DIM * SEQ;
  const unsigned char* mrow = mask + (size_t)bi * SEQ;

  // Q A-fragments, resident across the whole key loop (16x128 per wave)
  v16bf aq[4];
#pragma unroll
  for (int ks = 0; ks < 4; ++ks) aq[ks] = load_a_frag(Qb + ks * 32, HEAD_DIM);

  float mi[8], li[8];
#pragma unroll
  for (int r = 0; r < 8; ++r) { mi[r] = -3.0e38f; li[r] = 0.0f; }
  v8f o[8] = {};

  const int col    = lane & 15;
  const int rowoff = (lane >> 4) * 8;

  // loop-carried K-tile fragments
  v16bf bk[2][4];
#pragma unroll
  for (int nt = 0; nt < 2; ++nt)
#pragma unroll
    for (int ks = 0; ks < 4; ++ks)
      bk[nt][ks] = load_b_frag(Kb + (size_t)(nt * 16) * HEAD_DIM + ks * 32, HEAD_DIM);

  for (int kt = 0; kt < SEQ / 32; ++kt) {
    const int key0 = kt * 32;

    // ---- S = Q * K^T (16x32 per wave) : 8 WMMAs on pre-loaded fragments ----
    v8f sacc[2];
#pragma unroll
    for (int nt = 0; nt < 2; ++nt) {
      v8f s = {};
#pragma unroll
      for (int ks = 0; ks < 4; ++ks) s = wmma_bf16(aq[ks], bk[nt][ks], s);
      // key padding mask: column n of this tile == one key per lane
      if (mrow[key0 + nt * 16 + col]) {
#pragma unroll
        for (int r = 0; r < 8; ++r) s[r] = -1.0e30f;
      }
      sacc[nt] = s;
    }

    // ---- prefetch: this tile's V frags + next tile's K frags ----
    v16bf bv[8];
#pragma unroll
    for (int ot = 0; ot < 8; ++ot)
      bv[ot] = load_b_frag(Vb + (size_t)(ot * 16) * SEQ + key0, SEQ);
    if (kt + 1 < SEQ / 32) {
      const int keyn = key0 + 32;
#pragma unroll
      for (int nt = 0; nt < 2; ++nt)
#pragma unroll
        for (int ks = 0; ks < 4; ++ks)
          bk[nt][ks] = load_b_frag(Kb + (size_t)(keyn + nt * 16) * HEAD_DIM + ks * 32,
                                   HEAD_DIM);
    }

    // ---- online softmax: row stats via half-wave shfl reductions ----
#pragma unroll
    for (int r = 0; r < 8; ++r) {
      float rmax = fmaxf(sacc[0][r], sacc[1][r]);
      rmax = fmaxf(rmax, __shfl_xor(rmax, 1));
      rmax = fmaxf(rmax, __shfl_xor(rmax, 2));
      rmax = fmaxf(rmax, __shfl_xor(rmax, 4));
      rmax = fmaxf(rmax, __shfl_xor(rmax, 8));
      const float newm  = fmaxf(mi[r], rmax);
      const float alpha = __expf(mi[r] - newm);
      mi[r] = newm;
      const float p0 = __expf(sacc[0][r] - newm);
      const float p1 = __expf(sacc[1][r] - newm);
      sacc[0][r] = p0; sacc[1][r] = p1;
      float rsum = p0 + p1;
      rsum += __shfl_xor(rsum, 1);
      rsum += __shfl_xor(rsum, 2);
      rsum += __shfl_xor(rsum, 4);
      rsum += __shfl_xor(rsum, 8);
      li[r] = li[r] * alpha + rsum;
#pragma unroll
      for (int ot = 0; ot < 8; ++ot) o[ot][r] *= alpha;  // rescale O rows
    }

    // ---- P: C-layout regs -> per-wave LDS (row-major 16x32) -> A-fragment --
#pragma unroll
    for (int nt = 0; nt < 2; ++nt)
#pragma unroll
      for (int r = 0; r < 8; ++r)
        p_lds[wave][(rowoff + r) * 32 + nt * 16 + col] = (bf16)sacc[nt][r];
    __builtin_amdgcn_wave_barrier();   // order stores before the reload
    v16bf ap = load_a_frag(&p_lds[wave][0], 32);
    __builtin_amdgcn_wave_barrier();   // order reload before next tile's stores

    // ---- O += P * V (V fragments already in flight/registers) ----
#pragma unroll
    for (int ot = 0; ot < 8; ++ot) o[ot] = wmma_bf16(ap, bv[ot], o[ot]);
  }

  // ---- normalize by l and store to (b, t, D) bf16 ----
  float inv[8];
#pragma unroll
  for (int r = 0; r < 8; ++r) inv[r] = 1.0f / li[r];
#pragma unroll
  for (int ot = 0; ot < 8; ++ot) {
    const int d = ot * 16 + col;
#pragma unroll
    for (int r = 0; r < 8; ++r) {
      const int trow = q0 + rowoff + r;
      OUT[((size_t)bi * SEQ + trow) * D_MODEL + h * HEAD_DIM + d] =
          (bf16)(o[ot][r] * inv[r]);
    }
  }
}

// ===========================================================================
// Kernel 5: output projection  out[m, e] = sum_d attn[m, d] * w_out[e, d]
//   Same ping-pong 32x64-per-wave GEMM; fp32 written straight to d_out.
// ===========================================================================
__global__ void __launch_bounds__(256)
k_gemm_out(const bf16* __restrict__ A, const bf16* __restrict__ W,
           float* __restrict__ C) {
  const int wave = threadIdx.x >> 5;
  const int lane = threadIdx.x & 31;
  const int m0   = blockIdx.x * 64 + (wave >> 2) * 32;
  const int n0   = blockIdx.y * 256 + (wave & 3) * 64;

  const bf16* A0  = A + (size_t)m0 * D_MODEL;
  const bf16* A1  = A + (size_t)(m0 + 16) * D_MODEL;
  const bf16* Wn0 = W + (size_t)n0 * D_MODEL;

  v8f acc[2][4] = {};
  Frags ping, pong;
  load_frags(ping, A0, A1, Wn0, 0);

  int k = 0;
  for (; k + 64 < D_MODEL; k += 64) {
    load_frags(pong, A0, A1, Wn0, k + 32);
    do_wmmas(acc, ping);
    load_frags(ping, A0, A1, Wn0, k + 64);
    do_wmmas(acc, pong);
  }
  load_frags(pong, A0, A1, Wn0, k + 32);
  do_wmmas(acc, ping);
  do_wmmas(acc, pong);

  const int col    = lane & 15;
  const int rowoff = (lane >> 4) * 8;
#pragma unroll
  for (int mt = 0; mt < 2; ++mt) {
#pragma unroll
    for (int nt = 0; nt < 4; ++nt) {
      const int e = n0 + nt * 16 + col;
#pragma unroll
      for (int r = 0; r < 8; ++r) {
        const int m = m0 + mt * 16 + rowoff + r;
        C[(size_t)m * D_MODEL + e] = acc[mt][nt][r];
      }
    }
  }
}

// ===========================================================================
// Host launcher
// ===========================================================================
extern "C" void kernel_launch(void* const* d_in, const int* in_sizes, int n_in,
                              void* d_out, int out_size, void* d_ws, size_t ws_size,
                              hipStream_t stream) {
  (void)in_sizes; (void)n_in; (void)out_size; (void)ws_size;

  const float*         x     = (const float*)d_in[0];         // (4,2048,2048)
  const unsigned char* mask  = (const unsigned char*)d_in[1]; // (4,2048) bool
  const float*         w_qkv = (const float*)d_in[2];         // (6144,2048)
  const float*         w_out = (const float*)d_in[3];         // (2048,2048)
  float*               out   = (float*)d_out;                 // (4,2048,2048)

  // carve bf16 scratch from d_ws
  char*  ws  = (char*)d_ws;
  size_t off = 0;
  auto carve = [&](size_t bytes) -> void* {
    void* p = ws + off;
    off = (off + bytes + 255) & ~(size_t)255;
    return p;
  };
  const size_t NX = (size_t)BT * D_MODEL;        // 16,777,216
  const size_t NW = (size_t)QKV_N * D_MODEL;     // 12,582,912
  const size_t NO = (size_t)D_MODEL * D_MODEL;   //  4,194,304
  const size_t NH = (size_t)BATCH * N_HEADS * SEQ * HEAD_DIM; // 16,777,216

  bf16* xb  = (bf16*)carve(NX * 2);
  bf16* wqb = (bf16*)carve(NW * 2);
  bf16* wob = (bf16*)carve(NO * 2);
  bf16* qb  = (bf16*)carve(NH * 2);
  bf16* kb  = (bf16*)carve(NH * 2);
  bf16* vtb = (bf16*)carve(NH * 2);
  bf16* ab  = (bf16*)carve(NX * 2);

  k_cast8<<<(int)((NX / 8 + 255) / 256), 256, 0, stream>>>((const float4*)x,     (v8bf*)xb,  (int)(NX / 8));
  k_cast8<<<(int)((NW / 8 + 255) / 256), 256, 0, stream>>>((const float4*)w_qkv, (v8bf*)wqb, (int)(NW / 8));
  k_cast8<<<(int)((NO / 8 + 255) / 256), 256, 0, stream>>>((const float4*)w_out, (v8bf*)wob, (int)(NO / 8));

  k_gemm_qkv<<<dim3(BT / 64, QKV_N / 256), 256, 0, stream>>>(xb, wqb, qb, kb, vtb);

  k_rope<<<(BATCH * N_HEADS * SEQ) / 4, 256, 0, stream>>>(qb, kb);

  k_attn<<<dim3(SEQ / 64, BATCH * N_HEADS), 128, 0, stream>>>(qb, kb, vtb, mask, ab);

  k_gemm_out<<<dim3(BT / 64, D_MODEL / 256), 256, 0, stream>>>(ab, wob, out);
}
